// Predict_LoRA_Linear_80874234183963
// MI455X (gfx1250) — matrix operand order repaired and checked by reading, not compiled, not run on hardware
//
#include <hip/hip_runtime.h>

// ---------- types ----------
typedef __attribute__((ext_vector_type(16))) __bf16 v16bf;
typedef __attribute__((ext_vector_type(8)))  __bf16 v8bf;
typedef __attribute__((ext_vector_type(8)))  float  v8f;

// ---------- problem constants (fixed by reference) ----------
constexpr int B   = 4;
constexpr int S   = 2048;
constexpr int D   = 2048;   // K of main GEMM
constexpr int R   = 2048;   // N of main GEMM
constexpr int LR  = 16;     // LoRA rank
constexpr int NTOK = B * S; // 8192 = M of main GEMM
constexpr float SCALE = 1.0f;

// ---------- helpers ----------
static __device__ __forceinline__ unsigned short f32_to_bf16_rne(float f) {
    unsigned u = __builtin_bit_cast(unsigned, f);
    unsigned r = u + 0x7fffu + ((u >> 16) & 1u);   // round to nearest even
    return (unsigned short)(r >> 16);
}
static __device__ __forceinline__ float bf16_bits_to_f32(unsigned short h) {
    unsigned u = ((unsigned)h) << 16;
    return __builtin_bit_cast(float, u);
}

// ============================================================
// Kernel 0: split fp32 -> bf16 hi + bf16 lo (residual)
// ============================================================
__global__ __launch_bounds__(256)
void split_bf16_kernel(const float* __restrict__ src,
                       unsigned short* __restrict__ hi,
                       unsigned short* __restrict__ lo, int n) {
    int i = blockIdx.x * blockDim.x + threadIdx.x;
    if (i < n) {
        float f = src[i];
        unsigned short h = f32_to_bf16_rne(f);
        float rem = f - bf16_bits_to_f32(h);
        hi[i] = h;
        lo[i] = f32_to_bf16_rne(rem);
    }
}

// ============================================================
// Kernel 1: t[token,k] = sum_d input[token,d] * lora_right[b,d,k]
// one wave (32 lanes) per token, 8 waves per block
// ============================================================
__global__ __launch_bounds__(256)
void lora_t_kernel(const float* __restrict__ x,
                   const float* __restrict__ lright,
                   float* __restrict__ t) {
    __shared__ float red[8][32][LR + 1];   // +1 pad vs bank conflicts
    const int w    = threadIdx.x >> 5;
    const int lane = threadIdx.x & 31;
    const int token = blockIdx.x * 8 + w;
    const int b = token >> 11;             // S == 2048
    const float* xrow = x + (size_t)token * D;
    const float* rb   = lright + (size_t)b * D * LR;

    float acc[LR];
#pragma unroll
    for (int k = 0; k < LR; ++k) acc[k] = 0.0f;

    for (int d = lane; d < D; d += 32) {
        float xv = xrow[d];
        const float4* rp = (const float4*)(rb + (size_t)d * LR);
        float4 q0 = rp[0], q1 = rp[1], q2 = rp[2], q3 = rp[3];
        float r16[LR] = { q0.x,q0.y,q0.z,q0.w, q1.x,q1.y,q1.z,q1.w,
                          q2.x,q2.y,q2.z,q2.w, q3.x,q3.y,q3.z,q3.w };
#pragma unroll
        for (int k = 0; k < LR; ++k) acc[k] += xv * r16[k];
    }
#pragma unroll
    for (int k = 0; k < LR; ++k) red[w][lane][k] = acc[k];
    __syncthreads();
    if (lane < LR) {
        float s = 0.0f;
#pragma unroll
        for (int l = 0; l < 32; ++l) s += red[w][l][lane];
        t[(size_t)token * LR + lane] = s;
    }
}

// ============================================================
// Kernel 2: per-token  y = t @ lora_left, fold SCALE/||y|| into t (in place)
// one wave per token
// ============================================================
__global__ __launch_bounds__(256)
void lora_scale_kernel(const float* __restrict__ lleft,
                       float* __restrict__ t) {
    const int w    = threadIdx.x >> 5;
    const int lane = threadIdx.x & 31;
    const int token = blockIdx.x * 8 + w;
    const int b = token >> 11;
    const float* tb  = t + (size_t)token * LR;
    const float* llb = lleft + (size_t)b * LR * R;

    float tk[LR];
#pragma unroll
    for (int k = 0; k < LR; ++k) tk[k] = tb[k];

    float sumsq = 0.0f;
    for (int r = lane; r < R; r += 32) {
        float y = 0.0f;
#pragma unroll
        for (int k = 0; k < LR; ++k) y += tk[k] * llb[(size_t)k * R + r];
        sumsq += y * y;
    }
#pragma unroll
    for (int m = 16; m >= 1; m >>= 1) sumsq += __shfl_xor(sumsq, m, 32);

    float sc = SCALE * rsqrtf(sumsq);
    if (lane < LR) t[(size_t)token * LR + lane] = tb[lane] * sc;
}

// ============================================================
// Kernel 3: main GEMM via bf16x3 WMMA + fused epilogue
//   out[row,n] = sum_d A[row,d]*W[n,d] + bias[n] + sum_k t[row,k]*L[b,k,n]
// block = 256 threads = 8 waves, block tile 128(M) x 256(N)
// wave tile 64x64 = 4x4 tiles of 16x16, K step = 32
// per K step: 48 v_wmma vs 32 global_load_b128  (compute-bound by design)
// ============================================================
__global__ __launch_bounds__(256)
void gemm_wmma_kernel(const unsigned short* __restrict__ Ahi,
                      const unsigned short* __restrict__ Alo,
                      const unsigned short* __restrict__ Whi,
                      const unsigned short* __restrict__ Wlo,
                      const float* __restrict__ bias,
                      const float* __restrict__ lleft,
                      const float* __restrict__ ts,
                      float* __restrict__ out) {
    const int lane = threadIdx.x & 31;
    const int w    = threadIdx.x >> 5;
    const int wave_m = w & 1;        // 0..1
    const int wave_n = w >> 1;       // 0..3
    const int m0 = blockIdx.y * 128 + wave_m * 64;
    const int n0 = blockIdx.x * 256 + wave_n * 64;

    const int half = lane >> 4;      // 0: lanes 0-15, 1: lanes 16-31
    const int l15  = lane & 15;

    v8f c[4][4];
#pragma unroll
    for (int mt = 0; mt < 4; ++mt)
#pragma unroll
        for (int nt = 0; nt < 4; ++nt)
#pragma unroll
            for (int i = 0; i < 8; ++i) c[mt][nt][i] = 0.0f;

    for (int k0 = 0; k0 < D; k0 += 32) {
        // ---- B fragments (Kx16 = W rows along D, contiguous) ----
        // lanes 0-15: K = k0..k0+15 ; lanes 16-31: K = k0+16..k0+31
        v16bf bh[4], bl[4];
#pragma unroll
        for (int nt = 0; nt < 4; ++nt) {
            int n = n0 + nt * 16 + l15;
            size_t base = (size_t)n * D + (size_t)k0 + (size_t)half * 16;
            union { v16bf v; v8bf h[2]; } u;
            u.h[0] = *(const v8bf*)(Whi + base);
            u.h[1] = *(const v8bf*)(Whi + base + 8);
            bh[nt] = u.v;
            u.h[0] = *(const v8bf*)(Wlo + base);
            u.h[1] = *(const v8bf*)(Wlo + base + 8);
            bl[nt] = u.v;
        }
        // unconditional speculative prefetch of next K tile (no branches;
        // overruns stay inside the workspace / are dropped by HW)
        {
            size_t pb = (size_t)(n0 + l15) * D + (size_t)k0 + (size_t)half * 16;
            __builtin_prefetch(Whi + pb + 32, 0, 1);
        }
        // ---- A fragments + 48 WMMAs ----
#pragma unroll
        for (int mt = 0; mt < 4; ++mt) {
            int row = m0 + mt * 16 + l15;
            // lanes 0-15: elems 0-15 = K0-15
            // lanes 16-31: elems 0-15 = K16-31
            size_t ab = (size_t)row * D + (size_t)k0 + (size_t)half * 16;
            union { v16bf v; v8bf h[2]; } u;
            u.h[0] = *(const v8bf*)(Ahi + ab);
            u.h[1] = *(const v8bf*)(Ahi + ab + 8);
            v16bf ah = u.v;
            u.h[0] = *(const v8bf*)(Alo + ab);
            u.h[1] = *(const v8bf*)(Alo + ab + 8);
            v16bf al = u.v;
            if (mt == 0) __builtin_prefetch(Ahi + ab + 32, 0, 1);
#pragma unroll
            for (int nt = 0; nt < 4; ++nt) {
                c[mt][nt] = __builtin_amdgcn_wmma_f32_16x16x32_bf16(
                    false, ah, false, bh[nt], (short)0, c[mt][nt], false, false);
                c[mt][nt] = __builtin_amdgcn_wmma_f32_16x16x32_bf16(
                    false, ah, false, bl[nt], (short)0, c[mt][nt], false, false);
                c[mt][nt] = __builtin_amdgcn_wmma_f32_16x16x32_bf16(
                    false, al, false, bh[nt], (short)0, c[mt][nt], false, false);
            }
        }
    }

    // ---- epilogue: + bias + pre-scaled LoRA y ----
    // C layout: VGPR i -> M = i + 8*half, N = l15
    const int b = m0 >> 11;   // whole block in one batch (128 | 2048)
#pragma unroll
    for (int nt = 0; nt < 4; ++nt) {
        int n = n0 + nt * 16 + l15;
        float bn = bias[n];
        float Lc[LR];
        const float* llb = lleft + (size_t)b * LR * R + n;
#pragma unroll
        for (int k = 0; k < LR; ++k) Lc[k] = llb[(size_t)k * R];
#pragma unroll
        for (int mt = 0; mt < 4; ++mt) {
            int row_base = m0 + mt * 16 + half * 8;
#pragma unroll
            for (int i = 0; i < 8; ++i) {
                int row = row_base + i;
                const float4* tp = (const float4*)(ts + (size_t)row * LR);
                float4 q0 = tp[0], q1 = tp[1], q2 = tp[2], q3 = tp[3];
                float tt[LR] = { q0.x,q0.y,q0.z,q0.w, q1.x,q1.y,q1.z,q1.w,
                                 q2.x,q2.y,q2.z,q2.w, q3.x,q3.y,q3.z,q3.w };
                float y = 0.0f;
#pragma unroll
                for (int k = 0; k < LR; ++k) y += tt[k] * Lc[k];
                out[(size_t)row * R + n] = c[mt][nt][i] + bn + y;
            }
        }
    }
}

// ============================================================
// launcher
// ============================================================
extern "C" void kernel_launch(void* const* d_in, const int* in_sizes, int n_in,
                              void* d_out, int out_size, void* d_ws, size_t ws_size,
                              hipStream_t stream) {
    const float* inp    = (const float*)d_in[0];   // [B,S,D]
    const float* weight = (const float*)d_in[1];   // [R,D]
    const float* bias   = (const float*)d_in[2];   // [R]
    const float* lright = (const float*)d_in[3];   // [B,D,LR]
    const float* lleft  = (const float*)d_in[4];   // [B,LR,R]
    float* out = (float*)d_out;                    // [B,S,R]

    // workspace carve (bytes): Ahi|Alo|Whi|Wlo|t  ~= 84.4 MB
    unsigned short* Ahi = (unsigned short*)d_ws;
    unsigned short* Alo = Ahi + (size_t)NTOK * D;
    unsigned short* Whi = Alo + (size_t)NTOK * D;
    unsigned short* Wlo = Whi + (size_t)R * D;
    float*          t   = (float*)(Wlo + (size_t)R * D);

    const int nA = in_sizes[0];   // 16,777,216
    const int nW = in_sizes[1];   //  4,194,304
    split_bf16_kernel<<<(nA + 255) / 256, 256, 0, stream>>>(inp, Ahi, Alo, nA);
    split_bf16_kernel<<<(nW + 255) / 256, 256, 0, stream>>>(weight, Whi, Wlo, nW);

    lora_t_kernel<<<NTOK / 8, 256, 0, stream>>>(inp, lright, t);
    lora_scale_kernel<<<NTOK / 8, 256, 0, stream>>>(lleft, t);

    dim3 grid(R / 256, NTOK / 128);   // (8, 64)
    gemm_wmma_kernel<<<grid, 256, 0, stream>>>(Ahi, Alo, Whi, Wlo,
                                               bias, lleft, t, out);
}
